// MambaIntegerBlockV2Surprise_29815662969091
// MI455X (gfx1250) — compile-verified, matrix operand order checked by reading, not run on hardware
//
#include <hip/hip_runtime.h>
#include <hip/hip_bf16.h>
#include <math.h>

// ---------------- problem constants (from reference setup_inputs) ----------
#define BZ     4
#define LSEQ   2048
#define DMODEL 2048
#define NH     24
#define DH     64
#define NS     64
#define DI     1536          // NH*DH
#define N1     3072          // 2*DI
#define PREAL  3096          // H*(1+2S)
#define NPAD   3328          // PREAL padded to multiple of 256
#define MROWS  8192          // BZ*LSEQ

typedef _Float16 v8h  __attribute__((ext_vector_type(8)));
typedef _Float16 v16h __attribute__((ext_vector_type(16)));
typedef float    v8f  __attribute__((ext_vector_type(8)));

union V16U { v16h v; v8h h[2]; };

static __device__ __forceinline__ v16h ld_frag(const _Float16* p0, const _Float16* p1) {
    V16U u;
    u.h[0] = *reinterpret_cast<const v8h*>(p0);   // K = k0 + g*8 .. +7      (16B aligned)
    u.h[1] = *reinterpret_cast<const v8h*>(p1);   // K = k0 + 16 + g*8 .. +7 (16B aligned)
    return u.v;
}

// CDNA5 async DMA: global -> LDS, tracked by ASYNCcnt (no VGPR round trip).
static __device__ __forceinline__ void async_ld_b32(void* lds, const void* gaddr) {
    unsigned lo = (unsigned)(size_t)lds;                 // LDS_ADDR = flat[31:0]
    unsigned long long ga = (unsigned long long)(size_t)gaddr;
    asm volatile("global_load_async_to_lds_b32 %0, %1, off"
                 :: "v"(lo), "v"(ga) : "memory");
}
static __device__ __forceinline__ void wait_asynccnt0() {
    asm volatile("s_wait_asynccnt 0x0" ::: "memory");
}

// ---------------------------------------------------------------------------
// f16 WMMA GEMM:  out(M,N) = A(M,K) @ W(N,K)^T + bias, optional residual
// epilogue (mode 1): out = hs + gate*(acc + bias).
// Block = 256 threads = 8 waves (2 along M x 4 along N); block tile 64x256;
// wave tile 32x64: 2 A-frags + 4 B-frags -> 8 WMMAs per 32-K step
// (1.5 b128 loads per WMMA). Fragments loaded straight from row-major global
// memory using the ISA 16-bit 16x32 A-matrix lane layout (W^T symmetric).
// M mult of 64, N mult of 256, K mult of 32 -> no guards.
// ---------------------------------------------------------------------------
static __device__ __forceinline__ void store_tile(const v8f& acc, int mBase, int col,
    const float* __restrict__ bias, int nBias, float* __restrict__ out, int ldOut,
    int mode, const float* __restrict__ hs, float gv, int g)
{
    const float bv = (col < nBias) ? bias[col] : 0.f;
    #pragma unroll
    for (int r2 = 0; r2 < 8; ++r2) {
        const int row = mBase + g * 8 + r2;   // C/D layout: VGPR r2 -> M = 8*g + r2
        const size_t idx = (size_t)row * ldOut + col;
        float v = acc[r2] + bv;
        if (mode == 1) v = hs[idx] + gv * v;
        out[idx] = v;
    }
}

__global__ __launch_bounds__(256)
void wmma_gemm_kernel(const _Float16* __restrict__ A,
                      const _Float16* __restrict__ W,
                      const float* __restrict__ bias, int nBias,
                      float* __restrict__ out, int ldOut, int K,
                      int mode, const float* __restrict__ hs,
                      const float* __restrict__ gate)
{
    const int lane = threadIdx.x & 31;
    const int wave = threadIdx.x >> 5;
    const int wm   = wave & 1;           // 2 waves along M
    const int wn   = wave >> 1;          // 4 waves along N
    const int mBase = blockIdx.y * 64 + wm * 32;
    const int nBase = blockIdx.x * 256 + wn * 64;
    const int r = lane & 15;             // matrix row within fragment
    const int g = lane >> 4;             // lane half (selects K group)

    const _Float16* __restrict__ A0 = A + (size_t)(mBase + r) * K;
    const _Float16* __restrict__ A1 = A + (size_t)(mBase + 16 + r) * K;
    const _Float16* __restrict__ W0 = W + (size_t)(nBase + r) * K;
    const _Float16* __restrict__ W1 = W + (size_t)(nBase + 16 + r) * K;
    const _Float16* __restrict__ W2 = W + (size_t)(nBase + 32 + r) * K;
    const _Float16* __restrict__ W3 = W + (size_t)(nBase + 48 + r) * K;

    v8f acc00 = {0.f,0.f,0.f,0.f,0.f,0.f,0.f,0.f};
    v8f acc01 = acc00, acc02 = acc00, acc03 = acc00;
    v8f acc10 = acc00, acc11 = acc00, acc12 = acc00, acc13 = acc00;

    const int g8 = g * 8;
    for (int k0 = 0; k0 < K; k0 += 32) {
        const int off0 = k0 + g8;
        const int off1 = k0 + 16 + g8;
        __builtin_prefetch(A0 + off0 + 128, 0, 3);    // global_prefetch_b8
        __builtin_prefetch(W0 + off0 + 128, 0, 3);
        v16h a0 = ld_frag(A0 + off0, A0 + off1);
        v16h a1 = ld_frag(A1 + off0, A1 + off1);
        v16h b0 = ld_frag(W0 + off0, W0 + off1);
        v16h b1 = ld_frag(W1 + off0, W1 + off1);
        v16h b2 = ld_frag(W2 + off0, W2 + off1);
        v16h b3 = ld_frag(W3 + off0, W3 + off1);
        acc00 = __builtin_amdgcn_wmma_f32_16x16x32_f16(false, a0, false, b0, (short)0, acc00, false, false);
        acc01 = __builtin_amdgcn_wmma_f32_16x16x32_f16(false, a0, false, b1, (short)0, acc01, false, false);
        acc02 = __builtin_amdgcn_wmma_f32_16x16x32_f16(false, a0, false, b2, (short)0, acc02, false, false);
        acc03 = __builtin_amdgcn_wmma_f32_16x16x32_f16(false, a0, false, b3, (short)0, acc03, false, false);
        acc10 = __builtin_amdgcn_wmma_f32_16x16x32_f16(false, a1, false, b0, (short)0, acc10, false, false);
        acc11 = __builtin_amdgcn_wmma_f32_16x16x32_f16(false, a1, false, b1, (short)0, acc11, false, false);
        acc12 = __builtin_amdgcn_wmma_f32_16x16x32_f16(false, a1, false, b2, (short)0, acc12, false, false);
        acc13 = __builtin_amdgcn_wmma_f32_16x16x32_f16(false, a1, false, b3, (short)0, acc13, false, false);
    }

    const float gv = (mode == 1) ? gate[0] : 0.f;
    const int c0 = nBase + r, c1 = c0 + 16, c2 = c0 + 32, c3 = c0 + 48;
    store_tile(acc00, mBase,      c0, bias, nBias, out, ldOut, mode, hs, gv, g);
    store_tile(acc01, mBase,      c1, bias, nBias, out, ldOut, mode, hs, gv, g);
    store_tile(acc02, mBase,      c2, bias, nBias, out, ldOut, mode, hs, gv, g);
    store_tile(acc03, mBase,      c3, bias, nBias, out, ldOut, mode, hs, gv, g);
    store_tile(acc10, mBase + 16, c0, bias, nBias, out, ldOut, mode, hs, gv, g);
    store_tile(acc11, mBase + 16, c1, bias, nBias, out, ldOut, mode, hs, gv, g);
    store_tile(acc12, mBase + 16, c2, bias, nBias, out, ldOut, mode, hs, gv, g);
    store_tile(acc13, mBase + 16, c3, bias, nBias, out, ldOut, mode, hs, gv, g);
}

// ---------------- RMSNorm -> f16 activations -------------------------------
__global__ __launch_bounds__(256)
void rmsnorm_kernel(const float* __restrict__ hs, const float* __restrict__ w,
                    _Float16* __restrict__ xn)
{
    const int row = blockIdx.x;
    const float* rp = hs + (size_t)row * DMODEL;
    float ss = 0.f;
    for (int i = threadIdx.x; i < DMODEL; i += 256) { float v = rp[i]; ss += v * v; }
    __shared__ float sh[256];
    sh[threadIdx.x] = ss; __syncthreads();
    for (int st = 128; st > 0; st >>= 1) {
        if (threadIdx.x < st) sh[threadIdx.x] += sh[threadIdx.x + st];
        __syncthreads();
    }
    const float scale = rsqrtf(sh[0] * (1.f / DMODEL) + 1.1920928955078125e-07f);
    _Float16* op = xn + (size_t)row * DMODEL;
    for (int i = threadIdx.x; i < DMODEL; i += 256)
        op[i] = (_Float16)(rp[i] * scale * w[i]);
}

// ---------------- f32 -> f16 weight convert with zero tail padding ---------
__global__ void cvt_f32_to_f16_pad(const float* __restrict__ src,
                                   _Float16* __restrict__ dst,
                                   long nsrc, long ndst)
{
    long i = (long)blockIdx.x * 256 + threadIdx.x;
    const long stride = (long)gridDim.x * 256;
    for (; i < ndst; i += stride)
        dst[i] = (i < nsrc) ? (_Float16)src[i] : (_Float16)0.f;
}

// ---------------- depthwise causal conv(k=4) + SiLU ------------------------
__global__ void conv_silu_kernel(const float* __restrict__ xz,
                                 const float* __restrict__ cw,
                                 const float* __restrict__ cb,
                                 float* __restrict__ xcf,
                                 _Float16* __restrict__ xch)
{
    long i = (long)blockIdx.x * 256 + threadIdx.x;
    const long stride = (long)gridDim.x * 256;
    const long total = (long)MROWS * DI;
    for (; i < total; i += stride) {
        const int c   = (int)(i % DI);
        const int row = (int)(i / DI);
        const int l   = row & (LSEQ - 1);
        float acc = cb[c];
        #pragma unroll
        for (int t = 0; t < 4; ++t) {
            const int ll = l - 3 + t;
            if (ll >= 0)
                acc += cw[c * 4 + t] * xz[(size_t)(row - 3 + t) * N1 + c];
        }
        const float y = acc / (1.f + expf(-acc));   // silu
        xcf[i] = y;
        xch[i] = (_Float16)y;
    }
}

// ---------------- per (b,l,h): alpha, normalized B/C (scan-friendly layout)
// h_red in the reference carry is dead, so alpha depends only on (dt, xmean,
// Bn) of the current step and is fully precomputable.
__global__ __launch_bounds__(64)
void scanprep_kernel(const float* __restrict__ xp,
                     const float* __restrict__ xcf,
                     const float* __restrict__ A_log,
                     const float* __restrict__ l2ab,
                     const float* __restrict__ l2b,
                     const float* __restrict__ sema,
                     float* __restrict__ alphaO,
                     float* __restrict__ bnO,
                     float* __restrict__ cnO)
{
    const int l = blockIdx.x, h = blockIdx.y, b = blockIdx.z;
    const int s = threadIdx.x;                      // 0..63 (S == Dh == 64)
    const int row = b * LSEQ + l;
    const size_t xpbase = (size_t)row * NPAD + h * 129;
    const float Bp = xp[xpbase + 1 + s];
    const float Cp = xp[xpbase + 1 + NS + s];
    const float xv = xcf[(size_t)row * DI + h * DH + s];

    __shared__ float shx[64], shb[64], shc[64];
    shx[s] = xv; shb[s] = Bp * Bp; shc[s] = Cp * Cp;
    __syncthreads();
    for (int st = 32; st > 0; st >>= 1) {
        if (s < st) { shx[s] += shx[s+st]; shb[s] += shb[s+st]; shc[s] += shc[s+st]; }
        __syncthreads();
    }
    const float xmean = shx[0] * (1.f / DH);
    const float sb2 = shb[0], sc2 = shc[0];
    const float dB = fmaxf(sqrtf(sb2), 1e-12f);
    const float dC = fmaxf(sqrtf(sc2), 1e-12f);

    const size_t obase = (((size_t)b * NH + h) * LSEQ + l) * NS;
    bnO[obase + s] = Bp / dB;
    cnO[obase + s] = Cp / dC;

    if (s == 0) {
        const float dt = xp[xpbase];
        const float Ac = fminf(fmaxf(A_log[h], -2.3f), -0.01f);
        const float sp = (dt > 20.f) ? dt : log1pf(expf(dt));   // softplus
        const float a  = expf(Ac * sp);                         // A_decay
        const float ab = 1.f - exp2f(fminf(fmaxf(l2ab[h], -3.32f), -0.015f));
        const float be = exp2f(fminf(fmaxf(l2b[h], -2.f), 2.f));
        const float om = 1.f - a;
        const float pe = om * om * xmean * xmean * (sb2 / (dB * dB)) * (1.f / NS);
        const float ns = pe / (sema[h] + 1e-6f);
        const float boost = fmaxf(tanhf(be * ns), 0.f);
        const float alpha = fminf(fmaxf(ab + (1.f - ab) * boost, 0.01f), 0.999f);
        alphaO[((size_t)b * NH + h) * LSEQ + l] = alpha;
    }
}

// ---------------- sequential scan: h = a*h + (1-a)*B(x)x ; y = C^T h --------
// One block per (b,h). State lives in registers (thread t owns d=t&63 and 16
// s-values). Per-step inputs are double-buffered in LDS and fetched with
// CDNA5 async DMA (global_load_async_to_lds_b32 / s_wait_asynccnt): step l+1
// loads overlap step l compute.
__global__ __launch_bounds__(256)
void scan_kernel(const float* __restrict__ xcf,
                 const float* __restrict__ alphaA,
                 const float* __restrict__ bnA,
                 const float* __restrict__ cnA,
                 float* __restrict__ Y)
{
    const int bh = blockIdx.x;
    const int b = bh / NH, h = bh % NH;
    const int t  = threadIdx.x;
    const int d  = t & 63;
    const int sg = t >> 6;               // 0..3
    __shared__ float sx[2][64], sb[2][64], sc[2][64], sal[2];
    __shared__ float yp[256];

    float hreg[16];
    #pragma unroll
    for (int j = 0; j < 16; ++j) hreg[j] = 0.f;

    const size_t xrow0  = ((size_t)b * LSEQ) * DI + (size_t)h * DH;
    const size_t bcbase = ((size_t)b * NH + h) * LSEQ * NS;
    const size_t albase = ((size_t)b * NH + h) * LSEQ;

    auto issue = [&](int l, int buf) {
        if (t < 64)        async_ld_b32(&sx[buf][t],     xcf + xrow0 + (size_t)l * DI + t);
        else if (t < 128)  async_ld_b32(&sb[buf][t-64],  bnA + bcbase + (size_t)l * NS + (t - 64));
        else if (t < 192)  async_ld_b32(&sc[buf][t-128], cnA + bcbase + (size_t)l * NS + (t - 128));
        else if (t == 192) async_ld_b32(&sal[buf],       alphaA + albase + l);
    };

    issue(0, 0);
    wait_asynccnt0();
    __syncthreads();

    for (int l = 0; l < LSEQ; ++l) {
        const int buf = l & 1;
        if (l + 1 < LSEQ) issue(l + 1, buf ^ 1);   // overlap next step's DMA

        const float al = sal[buf], om = 1.f - al;
        const float xd = sx[buf][d];
        float acc = 0.f;
        #pragma unroll
        for (int j = 0; j < 16; ++j) {
            const int s = sg * 16 + j;
            hreg[j] = fmaf(al, hreg[j], om * sb[buf][s] * xd);
            acc = fmaf(sc[buf][s], hreg[j], acc);
        }
        yp[t] = acc;
        __syncthreads();

        if (t < 64)
            Y[xrow0 + (size_t)l * DI + t] = yp[t] + yp[t+64] + yp[t+128] + yp[t+192];

        wait_asynccnt0();    // publish buf^1 ...
        __syncthreads();     // ... to all waves before next iteration
    }
}

// ---------------- y * silu(z) -> f16 ---------------------------------------
__global__ void gate_kernel(const float* __restrict__ Y,
                            const float* __restrict__ xz,
                            _Float16* __restrict__ yg)
{
    long i = (long)blockIdx.x * 256 + threadIdx.x;
    const long stride = (long)gridDim.x * 256;
    const long total = (long)MROWS * DI;
    for (; i < total; i += stride) {
        const int c = (int)(i % DI);
        const long row = i / DI;
        const float z = xz[(size_t)row * N1 + DI + c];
        const float sz = z / (1.f + expf(-z));
        yg[i] = (_Float16)(Y[i] * sz);
    }
}

// ---------------------------------------------------------------------------
extern "C" void kernel_launch(void* const* d_in, const int* in_sizes, int n_in,
                              void* d_out, int out_size, void* d_ws, size_t ws_size,
                              hipStream_t stream)
{
    const float* hs    = (const float*)d_in[0];
    const float* normw = (const float*)d_in[1];
    const float* Win   = (const float*)d_in[2];
    const float* bin   = (const float*)d_in[3];
    const float* cw    = (const float*)d_in[4];
    const float* cb    = (const float*)d_in[5];
    const float* Wxp   = (const float*)d_in[6];
    const float* bxp   = (const float*)d_in[7];
    const float* A_log = (const float*)d_in[8];
    const float* l2ab  = (const float*)d_in[9];
    const float* l2b   = (const float*)d_in[10];
    const float* sema  = (const float*)d_in[11];
    const float* Wout  = (const float*)d_in[12];
    const float* bout  = (const float*)d_in[13];
    const float* rg    = (const float*)d_in[14];
    float* out = (float*)d_out;

    char* ws = (char*)d_ws;
    size_t off = 0;
    auto alloc = [&](size_t bytes) -> size_t {
        size_t o = off; off += (bytes + 255) & ~(size_t)255; return o;
    };

    _Float16* win16  = (_Float16*)(ws + alloc((size_t)N1 * DMODEL * 2));
    _Float16* wxp16  = (_Float16*)(ws + alloc((size_t)NPAD * DI * 2));
    _Float16* wout16 = (_Float16*)(ws + alloc((size_t)DMODEL * DI * 2));
    _Float16* xn16   = (_Float16*)(ws + alloc((size_t)MROWS * DMODEL * 2));
    float*    xz     = (float*)   (ws + alloc((size_t)MROWS * N1 * 4));
    float*    xcf    = (float*)   (ws + alloc((size_t)MROWS * DI * 4));
    _Float16* xch    = (_Float16*)(ws + alloc((size_t)MROWS * DI * 2));
    float*    xp     = (float*)   (ws + alloc((size_t)MROWS * NPAD * 4));
    float*    alphaB = (float*)   (ws + alloc((size_t)BZ * NH * LSEQ * 4));
    float*    bnB    = (float*)   (ws + alloc((size_t)BZ * NH * LSEQ * NS * 4));
    float*    cnB    = (float*)   (ws + alloc((size_t)BZ * NH * LSEQ * NS * 4));
    float*    Ybuf   = xp;     // alias: xp fully consumed by scanprep before scan writes Y
    _Float16* yg16   = xn16;   // alias: xn16 fully consumed by GEMM1 before gate writes

    // 1) weight conversions (x_proj_w zero-padded 3096 -> 3328 rows)
    cvt_f32_to_f16_pad<<<2048, 256, 0, stream>>>(Win,  win16,  (long)N1 * DMODEL,  (long)N1 * DMODEL);
    cvt_f32_to_f16_pad<<<2048, 256, 0, stream>>>(Wxp,  wxp16,  (long)PREAL * DI,   (long)NPAD * DI);
    cvt_f32_to_f16_pad<<<2048, 256, 0, stream>>>(Wout, wout16, (long)DMODEL * DI,  (long)DMODEL * DI);

    // 2) RMSNorm -> f16
    rmsnorm_kernel<<<MROWS, 256, 0, stream>>>(hs, normw, xn16);

    // 3) in_proj GEMM: xz = xn @ Win^T + bin   (8192 x 3072, K=2048)
    wmma_gemm_kernel<<<dim3(N1 / 256, MROWS / 64), 256, 0, stream>>>(
        xn16, win16, bin, N1, xz, N1, DMODEL, 0, nullptr, nullptr);

    // 4) depthwise conv + silu
    conv_silu_kernel<<<4096, 256, 0, stream>>>(xz, cw, cb, xcf, xch);

    // 5) x_proj GEMM: xp = xconv @ Wxp^T + bxp  (8192 x 3328pad, K=1536)
    wmma_gemm_kernel<<<dim3(NPAD / 256, MROWS / 64), 256, 0, stream>>>(
        xch, wxp16, bxp, PREAL, xp, NPAD, DI, 0, nullptr, nullptr);

    // 6) alpha / Bn / Cn precompute (scan-friendly (b,h,l,.) layout)
    scanprep_kernel<<<dim3(LSEQ, NH, BZ), 64, 0, stream>>>(
        xp, xcf, A_log, l2ab, l2b, sema, alphaB, bnB, cnB);

    // 7) sequential scan, one block per (b,h), state in registers,
    //    async-DMA double-buffered LDS inputs
    scan_kernel<<<BZ * NH, 256, 0, stream>>>(xcf, alphaB, bnB, cnB, Ybuf);

    // 8) Y * silu(z) -> f16
    gate_kernel<<<4096, 256, 0, stream>>>(Ybuf, xz, yg16);

    // 9) out_proj GEMM + fused bias/res_gate/residual epilogue -> d_out
    wmma_gemm_kernel<<<dim3(DMODEL / 256, MROWS / 64), 256, 0, stream>>>(
        yg16, wout16, bout, DMODEL, out, DMODEL, DI, 1, hs, rg);

    (void)in_sizes; (void)n_in; (void)out_size; (void)ws_size;
}
